// MultiheadSdpaDiff1_21208548507894
// MI455X (gfx1250) — matrix operand order repaired and checked by reading, not compiled, not run on hardware
//
#include <hip/hip_runtime.h>

// ---------------------------------------------------------------------------
// MI455X (gfx1250) differential attention, bf16 WMMA pipeline.
// S=2048, E=2048, B=2, H=8 head-pairs, D=128. wave32.
// Round 3: software-pipeline gemm_tile (double-buffered fragments) so WMMAs
// wait on loads issued one iteration earlier -> staggered s_wait_loadcnt.
// ---------------------------------------------------------------------------

typedef __attribute__((ext_vector_type(16))) __bf16 bf16x16;
typedef __attribute__((ext_vector_type(8)))  float  f32x8;

union Frag { bf16x16 v; uint4 u[2]; };

#define WMMA_BF16(a, b, c) \
  __builtin_amdgcn_wmma_f32_16x16x32_bf16(false, (a), false, (b), (short)0, (c), false, false)

__device__ __forceinline__ unsigned short f2bf(float f) {
  union { float f; unsigned u; } v; v.f = f;
  unsigned r = v.u + 0x7FFFu + ((v.u >> 16) & 1u);   // round-to-nearest-even
  return (unsigned short)(r >> 16);
}

static constexpr int S = 2048;
static constexpr int E = 2048;
static constexpr int D = 128;
static constexpr int H = 8;               // head pairs
static constexpr float SCALE = 0.08838834764831845f;       // D^-0.5
static constexpr float LAMBDA_INIT = 0.78360576653162443f; // 0.8-0.6*exp(-3.6)

// ---------------------------------------------------------------------------
// 1) fp32 -> bf16 convert (x). Each thread converts 4 elements.
// ---------------------------------------------------------------------------
__global__ __launch_bounds__(256) void cvt_f32_bf16(const float* __restrict__ src,
                                                    unsigned short* __restrict__ dst) {
  int i = blockIdx.x * blockDim.x + threadIdx.x;
  float4 v = reinterpret_cast<const float4*>(src)[i];
  union { unsigned short s[4]; uint2 u; } o;
  o.s[0] = f2bf(v.x); o.s[1] = f2bf(v.y); o.s[2] = f2bf(v.z); o.s[3] = f2bf(v.w);
  reinterpret_cast<uint2*>(dst)[i] = o.u;
}

// ---------------------------------------------------------------------------
// 2) Pack W[K=2048, N=2048] (row-major fp32) into this file's B-fragment order (contiguous k, same as ld_a):
//    frag f = n_tile*64 + k32tile, 32 lanes x 16 bf16 contiguous per lane.
//    Lane L: column n = n_tile*16 + (L&15); K elems = k32*32 + (L>>4)*16 + 0..15
// ---------------------------------------------------------------------------
__global__ __launch_bounds__(256) void pack_w(const float* __restrict__ W,
                                              unsigned short* __restrict__ out) {
  const int w = threadIdx.x >> 5, lane = threadIdx.x & 31;
  const int hi = lane >> 4, lr = lane & 15;
  const int f = blockIdx.x * 8 + w;       // 8192 fragments
  const int ntile = f >> 6;
  const int kt = f & 63;
  const int n = ntile * 16 + lr;
  union { unsigned short s[16]; uint4 u[2]; } o;
#pragma unroll
  for (int e = 0; e < 16; ++e) {
    int k = kt * 32 + hi * 16 + e;
    o.s[e] = f2bf(W[(size_t)k * E + n]);
  }
  uint4* dst = reinterpret_cast<uint4*>(out + (size_t)f * 512 + lane * 16);
  dst[0] = o.u[0]; dst[1] = o.u[1];
}

// ---------------------------------------------------------------------------
// GEMM core: C[mbase..+16, nbase..+64] += A(bf16 row-major, lda=2048) x Bfrag
// K = 2048 in 64 steps of 32, double-buffered: the k+1 fragment set is loaded
// before the k WMMA chain runs, so each v_wmma waits only until the older
// loads land (s_wait_loadcnt ~0xa) while newer loads stay in flight.
// ---------------------------------------------------------------------------
__device__ __forceinline__ void ld_a(Frag& a, const unsigned short* arow, int k32, int hi) {
  const unsigned short* ap = arow + k32 * 32 + hi * 16;  // K {0..15}/{16..31}
  a.u[0] = *reinterpret_cast<const uint4*>(ap);
  a.u[1] = *reinterpret_cast<const uint4*>(ap + 8);
}
__device__ __forceinline__ void ld_b(Frag b[4], const unsigned short* bcol, int k32) {
#pragma unroll
  for (int nt = 0; nt < 4; ++nt) {
    const unsigned short* bp = bcol + (size_t)(nt * 64 + k32) * 512;
    b[nt].u[0] = *reinterpret_cast<const uint4*>(bp);
    b[nt].u[1] = *reinterpret_cast<const uint4*>(bp + 8);
  }
}

__device__ __forceinline__ void gemm_tile(const unsigned short* __restrict__ A,
                                          const unsigned short* __restrict__ Bf,
                                          int mbase, int nbase, f32x8 acc[4]) {
  const int lane = threadIdx.x & 31;
  const int hi = lane >> 4, lr = lane & 15;
  const unsigned short* arow = A + (size_t)(mbase + lr) * E;
  const unsigned short* bcol = Bf + ((size_t)(nbase >> 4) * 64 * 512) + lane * 16;

  Frag a0, a1, b0[4], b1[4];
  ld_a(a0, arow, 0, hi);
  ld_b(b0, bcol, 0);
  for (int k32 = 0; k32 < 62; k32 += 2) {
    ld_a(a1, arow, k32 + 1, hi);
    ld_b(b1, bcol, k32 + 1);
#pragma unroll
    for (int nt = 0; nt < 4; ++nt) acc[nt] = WMMA_BF16(a0.v, b0[nt].v, acc[nt]);
    ld_a(a0, arow, k32 + 2, hi);
    ld_b(b0, bcol, k32 + 2);
#pragma unroll
    for (int nt = 0; nt < 4; ++nt) acc[nt] = WMMA_BF16(a1.v, b1[nt].v, acc[nt]);
  }
  // k32 == 62 set already in (a0,b0); load 63, then drain both.
  ld_a(a1, arow, 63, hi);
  ld_b(b1, bcol, 63);
#pragma unroll
  for (int nt = 0; nt < 4; ++nt) acc[nt] = WMMA_BF16(a0.v, b0[nt].v, acc[nt]);
#pragma unroll
  for (int nt = 0; nt < 4; ++nt) acc[nt] = WMMA_BF16(a1.v, b1[nt].v, acc[nt]);
}

// ---------------------------------------------------------------------------
// 3) Fused Q/K/V projection. z=0 -> Q[b,h,comp,S,D], z=1 -> K (same layout),
//    z=2 -> Vt[b,h,2D,S] (transposed so PV B-fragments are contiguous).
// ---------------------------------------------------------------------------
__global__ __launch_bounds__(256) void proj_kernel(const unsigned short* __restrict__ xbf,
                                                   const unsigned short* __restrict__ wqf,
                                                   const unsigned short* __restrict__ wkf,
                                                   const unsigned short* __restrict__ wvf,
                                                   unsigned short* __restrict__ qbuf,
                                                   unsigned short* __restrict__ kbuf,
                                                   unsigned short* __restrict__ vtbuf) {
  const int w = threadIdx.x >> 5;
  const int lane = threadIdx.x & 31, hi = lane >> 4, lr = lane & 15;
  const int mbase = blockIdx.y * 64 + (w >> 1) * 16;
  const int nbase = blockIdx.x * 128 + (w & 1) * 64;
  const unsigned short* W = (blockIdx.z == 0) ? wqf : (blockIdx.z == 1) ? wkf : wvf;

  f32x8 acc[4];
#pragma unroll
  for (int nt = 0; nt < 4; ++nt)
#pragma unroll
    for (int r = 0; r < 8; ++r) acc[nt][r] = 0.f;

  gemm_tile(xbf, W, mbase, nbase, acc);

  if (blockIdx.z < 2) {
    unsigned short* outb = (blockIdx.z == 0) ? qbuf : kbuf;
#pragma unroll
    for (int nt = 0; nt < 4; ++nt) {
      const int n = nbase + nt * 16 + lr;
      const int h = n >> 8, comp = (n >> 7) & 1, d = n & 127;
#pragma unroll
      for (int r = 0; r < 8; ++r) {
        const int m = mbase + r + 8 * hi;
        const int b = m >> 11, s = m & 2047;
        outb[(((size_t)((b * H + h) * 2 + comp) * S + s) << 7) + d] = f2bf(acc[nt][r]);
      }
    }
  } else {
#pragma unroll
    for (int nt = 0; nt < 4; ++nt) {
      const int n = nbase + nt * 16 + lr;
      const int h = n >> 8, cd = n & 255;
      const int m0 = mbase + 8 * hi;
      const int b = m0 >> 11, s = m0 & 2047;
      union { unsigned short sh[8]; uint4 u; } pk;
#pragma unroll
      for (int r = 0; r < 8; ++r) pk.sh[r] = f2bf(acc[nt][r]);
      *reinterpret_cast<uint4*>(vtbuf + (((size_t)(b * H + h) * 256 + cd) << 11) + s) = pk.u;
    }
  }
}

// ---------------------------------------------------------------------------
// 4) Differential flash attention. Block = 8 waves: wave w -> branch comp=w&1,
//    query sub-tile qt=w>>1 (16 rows). Online softmax in WMMA C/D layout.
// ---------------------------------------------------------------------------
__global__ __launch_bounds__(256) void diffattn_kernel(
    const unsigned short* __restrict__ qbuf, const unsigned short* __restrict__ kbuf,
    const unsigned short* __restrict__ vtbuf,
    const float* __restrict__ lq1, const float* __restrict__ lk1,
    const float* __restrict__ lq2, const float* __restrict__ lk2,
    const float* __restrict__ subln, unsigned short* __restrict__ abuf) {
  extern __shared__ char smem[];
  float* o2sh = reinterpret_cast<float*>(smem);                       // 4 * 16*256 f32
  unsigned short* pshm = reinterpret_cast<unsigned short*>(smem + 4 * 16 * 256 * 4);

  const int tid = threadIdx.x;
  const int w = tid >> 5, lane = tid & 31, hi = lane >> 4, lr = lane & 15;
  const int comp = w & 1, qt = w >> 1;
  const int qblock = blockIdx.x, h = blockIdx.y, b = blockIdx.z;
  const int qr0 = qblock * 64 + qt * 16;

  const size_t qk_off = ((size_t)((b * H + h) * 2 + comp)) * S * D;
  const unsigned short* Q = qbuf + qk_off;
  const unsigned short* K = kbuf + qk_off;
  const unsigned short* Vt = vtbuf + (size_t)(b * H + h) * 256 * S;

  // persistent Q A-fragments: 4 x (16x32) over D=128
  Frag qf[4];
  const unsigned short* qrow = Q + (size_t)(qr0 + lr) * D;
#pragma unroll
  for (int dc = 0; dc < 4; ++dc) {
    const unsigned short* p = qrow + dc * 32 + hi * 16;
    qf[dc].u[0] = *reinterpret_cast<const uint4*>(p);
    qf[dc].u[1] = *reinterpret_cast<const uint4*>(p + 8);
  }

  f32x8 o[16];
#pragma unroll
  for (int j = 0; j < 16; ++j)
#pragma unroll
    for (int r = 0; r < 8; ++r) o[j][r] = 0.f;
  float mrow[8], lrow[8];
#pragma unroll
  for (int r = 0; r < 8; ++r) { mrow[r] = -1e30f; lrow[r] = 0.f; }

  unsigned short* mysh = pshm + w * 512;       // 16 rows x 32 bf16
  const int kbmax = (qr0 + 15) >> 5;

  for (int kb = 0; kb <= kbmax; ++kb) {
    const unsigned short* kbase = K + (size_t)(kb * 32) * D;
    // --- S = Q K^T: batch 4 K-fragment loads, then 4-WMMA chain, twice ---
    f32x8 s0, s1;
#pragma unroll
    for (int r = 0; r < 8; ++r) { s0[r] = 0.f; s1[r] = 0.f; }
#pragma unroll
    for (int half = 0; half < 2; ++half) {
      Frag kf[4];
#pragma unroll
      for (int i = 0; i < 4; ++i) {          // i = {dc_lo, dc_hi} x {tile0, tile1}
        const int dc = half * 2 + (i >> 1);
        const int t2 = i & 1;
        const unsigned short* kp = kbase + (size_t)(t2 * 16 + lr) * D + dc * 32 + hi * 16;
        kf[i].u[0] = *reinterpret_cast<const uint4*>(kp);
        kf[i].u[1] = *reinterpret_cast<const uint4*>(kp + 8);
      }
      s0 = WMMA_BF16(qf[half * 2 + 0].v, kf[0].v, s0);
      s1 = WMMA_BF16(qf[half * 2 + 0].v, kf[1].v, s1);
      s0 = WMMA_BF16(qf[half * 2 + 1].v, kf[2].v, s0);
      s1 = WMMA_BF16(qf[half * 2 + 1].v, kf[3].v, s1);
    }
    // --- scale + causal mask (only last diagonal block needs it) ---
    const bool domask = (kb == kbmax);
#pragma unroll
    for (int r = 0; r < 8; ++r) {
      s0[r] *= SCALE; s1[r] *= SCALE;
      if (domask) {
        const int gq = qr0 + r + 8 * hi;
        const int gk = kb * 32 + lr;
        if (gk > gq) s0[r] = -1e30f;
        if (gk + 16 > gq) s1[r] = -1e30f;
      }
    }
    // --- online softmax: row max/sum over 16-lane halves via shfl_xor ---
    float mnew[8], cf[8];
#pragma unroll
    for (int r = 0; r < 8; ++r) {
      float t = fmaxf(s0[r], s1[r]);
      t = fmaxf(t, __shfl_xor(t, 1)); t = fmaxf(t, __shfl_xor(t, 2));
      t = fmaxf(t, __shfl_xor(t, 4)); t = fmaxf(t, __shfl_xor(t, 8));
      mnew[r] = fmaxf(mrow[r], t);
      cf[r] = __expf(mrow[r] - mnew[r]);
      mrow[r] = mnew[r];
    }
#pragma unroll
    for (int r = 0; r < 8; ++r) {
      s0[r] = __expf(s0[r] - mnew[r]);
      s1[r] = __expf(s1[r] - mnew[r]);
      float rs = s0[r] + s1[r];
      rs += __shfl_xor(rs, 1); rs += __shfl_xor(rs, 2);
      rs += __shfl_xor(rs, 4); rs += __shfl_xor(rs, 8);
      lrow[r] = lrow[r] * cf[r] + rs;
    }
#pragma unroll
    for (int j = 0; j < 16; ++j)
#pragma unroll
      for (int r = 0; r < 8; ++r) o[j][r] *= cf[r];
    // --- P: C/D layout -> LDS (row-major 16x32 bf16) -> A-fragment ---
#pragma unroll
    for (int r = 0; r < 8; ++r) {
      const int M = r + 8 * hi;
      mysh[M * 32 + lr]      = f2bf(s0[r]);
      mysh[M * 32 + 16 + lr] = f2bf(s1[r]);
    }
    asm volatile("s_wait_dscnt 0x0" ::: "memory");   // same-wave DS RAW; pin compiler
    Frag pf;
    {
      const unsigned short* pp = mysh + lr * 32 + hi * 16;
      pf.u[0] = *reinterpret_cast<const uint4*>(pp);
      pf.u[1] = *reinterpret_cast<const uint4*>(pp + 8);
    }
    // --- O += P @ Vt: 16 column tiles, V-fragments batched 4 at a time ---
    const unsigned short* vbase = Vt + (size_t)kb * 32 + hi * 16;
#pragma unroll
    for (int jg = 0; jg < 4; ++jg) {
      Frag vf[4];
#pragma unroll
      for (int i = 0; i < 4; ++i) {
        const unsigned short* vp = vbase + (size_t)((jg * 4 + i) * 16 + lr) * S;
        vf[i].u[0] = *reinterpret_cast<const uint4*>(vp);
        vf[i].u[1] = *reinterpret_cast<const uint4*>(vp + 8);
      }
#pragma unroll
      for (int i = 0; i < 4; ++i)
        o[jg * 4 + i] = WMMA_BF16(pf.v, vf[i].v, o[jg * 4 + i]);
    }
  }

  // softmax denominator
  float invl[8];
#pragma unroll
  for (int r = 0; r < 8; ++r) invl[r] = 1.f / lrow[r];
#pragma unroll
  for (int j = 0; j < 16; ++j)
#pragma unroll
    for (int r = 0; r < 8; ++r) o[j][r] *= invl[r];

  // branch 2 -> LDS
  if (comp == 1) {
    float* dst = o2sh + (size_t)qt * 16 * 256;
#pragma unroll
    for (int j = 0; j < 16; ++j)
#pragma unroll
      for (int r = 0; r < 8; ++r)
        dst[(r + 8 * hi) * 256 + j * 16 + lr] = o[j][r];
  }
  __syncthreads();

  // branch 1 waves: diff, RMSNorm(256), write bf16 activation
  if (comp == 0) {
    float d1 = 0.f, d2 = 0.f;
    for (int i = 0; i < D; ++i) { d1 += lq1[i] * lk1[i]; d2 += lq2[i] * lk2[i]; }
    const float lam = __expf(d1) - __expf(d2) + LAMBDA_INIT;

    const float* src = o2sh + (size_t)qt * 16 * 256;
    float ssq[8];
#pragma unroll
    for (int r = 0; r < 8; ++r) ssq[r] = 0.f;
#pragma unroll
    for (int j = 0; j < 16; ++j)
#pragma unroll
      for (int r = 0; r < 8; ++r) {
        float av = o[j][r] - lam * src[(r + 8 * hi) * 256 + j * 16 + lr];
        o[j][r] = av;
        ssq[r] += av * av;
      }
    float sc[8];
#pragma unroll
    for (int r = 0; r < 8; ++r) {
      float t = ssq[r];
      t += __shfl_xor(t, 1); t += __shfl_xor(t, 2);
      t += __shfl_xor(t, 4); t += __shfl_xor(t, 8);
      sc[r] = rsqrtf(t * (1.f / 256.f) + 1e-5f) * (1.f - LAMBDA_INIT);
    }
#pragma unroll
    for (int j = 0; j < 16; ++j) {
      const float sw = subln[j * 16 + lr];
#pragma unroll
      for (int r = 0; r < 8; ++r) {
        const int M = r + 8 * hi;
        const size_t row = (size_t)b * S + qr0 + M;
        abuf[row * E + h * 256 + j * 16 + lr] = f2bf(o[j][r] * sc[r] * sw);
      }
    }
  }
}

// ---------------------------------------------------------------------------
// 5) out = a @ Wo, fp32 row-major result.
// ---------------------------------------------------------------------------
__global__ __launch_bounds__(256) void out_gemm(const unsigned short* __restrict__ abuf,
                                                const unsigned short* __restrict__ wof,
                                                float* __restrict__ out) {
  const int w = threadIdx.x >> 5;
  const int lane = threadIdx.x & 31, hi = lane >> 4, lr = lane & 15;
  const int mbase = blockIdx.y * 64 + (w >> 1) * 16;
  const int nbase = blockIdx.x * 128 + (w & 1) * 64;
  f32x8 acc[4];
#pragma unroll
  for (int nt = 0; nt < 4; ++nt)
#pragma unroll
    for (int r = 0; r < 8; ++r) acc[nt][r] = 0.f;
  gemm_tile(abuf, wof, mbase, nbase, acc);
#pragma unroll
  for (int nt = 0; nt < 4; ++nt)
#pragma unroll
    for (int r = 0; r < 8; ++r)
      out[(size_t)(mbase + r + 8 * hi) * E + nbase + nt * 16 + lr] = acc[nt][r];
}

// ---------------------------------------------------------------------------
extern "C" void kernel_launch(void* const* d_in, const int* in_sizes, int n_in,
                              void* d_out, int out_size, void* d_ws, size_t ws_size,
                              hipStream_t stream) {
  const float* x    = (const float*)d_in[0];
  const float* Wq   = (const float*)d_in[1];
  const float* Wk   = (const float*)d_in[2];
  const float* Wv   = (const float*)d_in[3];
  const float* Wo   = (const float*)d_in[4];
  const float* lq1  = (const float*)d_in[5];
  const float* lk1  = (const float*)d_in[6];
  const float* lq2  = (const float*)d_in[7];
  const float* lk2  = (const float*)d_in[8];
  const float* subw = (const float*)d_in[9];
  float* out = (float*)d_out;

  char* ws = (char*)d_ws;
  size_t off = 0;
  auto take = [&](size_t bytes) { char* p = ws + off; off += bytes; return p; };
  unsigned short* xbf  = (unsigned short*)take((size_t)2 * S * E * 2);      // 16 MB
  unsigned short* wqf  = (unsigned short*)take((size_t)E * E * 2);          // 8 MB each
  unsigned short* wkf  = (unsigned short*)take((size_t)E * E * 2);
  unsigned short* wvf  = (unsigned short*)take((size_t)E * E * 2);
  unsigned short* wof  = (unsigned short*)take((size_t)E * E * 2);
  unsigned short* qbuf = (unsigned short*)take((size_t)2 * S * E * 2);      // 16 MB
  unsigned short* kbuf = (unsigned short*)take((size_t)2 * S * E * 2);
  unsigned short* vtbf = (unsigned short*)take((size_t)2 * S * E * 2);
  unsigned short* abuf = (unsigned short*)take((size_t)2 * S * E * 2);

  cvt_f32_bf16<<<8192, 256, 0, stream>>>(x, xbf);
  pack_w<<<1024, 256, 0, stream>>>(Wq, wqf);
  pack_w<<<1024, 256, 0, stream>>>(Wk, wkf);
  pack_w<<<1024, 256, 0, stream>>>(Wv, wvf);
  pack_w<<<1024, 256, 0, stream>>>(Wo, wof);
  proj_kernel<<<dim3(16, 64, 3), 256, 0, stream>>>(xbf, wqf, wkf, wvf, qbuf, kbuf, vtbf);
  const size_t shmem = 4 * 16 * 256 * sizeof(float) + 8 * 512 * sizeof(unsigned short);
  diffattn_kernel<<<dim3(32, 8, 2), 256, shmem, stream>>>(qbuf, kbuf, vtbf,
                                                          lq1, lk1, lq2, lk2, subw, abuf);
  out_gemm<<<dim3(16, 64), 256, 0, stream>>>(abuf, wof, out);
}
